// basket_GRU_MI_E2_76295799046568
// MI455X (gfx1250) — matrix operand order repaired and checked by reading, not compiled, not run on hardware
//
#include <hip/hip_runtime.h>
#include <hip/hip_bf16.h>
#include <math.h>

// Problem constants (from reference)
#define B_BATCH 64
#define T_SEQ   128
#define DIN     2048
#define HDIM    1024
#define DDISC   512
#define MROWS   (B_BATCH * T_SEQ)   // 8192

// GEMM tiling: 256 threads = 8 wave32, tile 64(M) x 128(N), K-step 32
#define BM 64
#define BN 128
#define BK 32
#define BKP 40            // padded LDS row stride (ushort): 80B, bank-spread, 16B aligned
#define NWG_SCAN 16       // persistent scan workgroups (co-resident)

typedef __bf16 bf16_t;
typedef bf16_t v16bf __attribute__((ext_vector_type(16)));
typedef float  v8f   __attribute__((ext_vector_type(8)));

union FragU { uint4 q[2]; v16bf v; };

__device__ __forceinline__ unsigned short f2bf(float f) {
  union { float f; unsigned u; } x; x.f = f;
  unsigned r = x.u + 0x7FFFu + ((x.u >> 16) & 1u);   // round-to-nearest-even
  return (unsigned short)(r >> 16);
}
__device__ __forceinline__ float sigmoidf_(float x) { return 1.f / (1.f + __expf(-x)); }

// ---------------------------------------------------------------------------
// Shared WMMA GEMM core: C[BMxBN] += A[BMxK] * B[KxN-slice], bf16 in, f32 acc.
// A row-major (lda), B row-major (ldb), B columns [bcol0, bcol0+BN).
// Wave w: M-tile = w>>1, N-half = (w&1)*64, 4 N-tiles of 16 -> acc[4] (v8f).
// ---------------------------------------------------------------------------
__device__ __forceinline__ void gemm_core(
    const unsigned short* __restrict__ A, int lda, int arow0,
    const unsigned short* __restrict__ Bm, int ldb, int bcol0,
    int K, v8f acc[4])
{
  __shared__ __align__(16) unsigned short sA[BM * BKP];    // padded [64][40]
  __shared__ __align__(16) unsigned short sBT[BN * BKP];   // transposed padded [n][k]

  const int tid  = threadIdx.x;
  const int lane = tid & 31;
  const int wid  = tid >> 5;
  const int wm   = wid >> 1;   // 0..3  (M-tile)
  const int wn   = wid & 1;    // 0..1  (64-col half)

  v8f z = {};
  acc[0] = z; acc[1] = z; acc[2] = z; acc[3] = z;

  const int ar  = tid >> 2;          // 0..63 row of A tile
  const int ac  = (tid & 3) * 8;     // 0/8/16/24 col (8 ushort = 16B)
  const int bn  = tid & 127;         // B column within tile
  const int bkh = (tid >> 7) * 16;   // K half 0/16

  for (int k0 = 0; k0 < K; k0 += BK) {
    // Stage A tile: each thread one 16B chunk (coalesced)
    *(uint4*)&sA[ar * BKP + ac] =
        *(const uint4*)&A[(size_t)(arow0 + ar) * lda + k0 + ac];
    // Stage B tile transposed: sBT[n][k] = B[k0+k][bcol0+n]
    // (for fixed kk, consecutive threads read consecutive columns -> coalesced)
    {
      const unsigned short* src = &Bm[(size_t)(k0 + bkh) * ldb + bcol0 + bn];
      unsigned short* dst = &sBT[bn * BKP + bkh];
#pragma unroll
      for (int kk = 0; kk < 16; ++kk) dst[kk] = src[(size_t)kk * ldb];
    }
    // Prefetch next K-tile into cache (global_prefetch_b8)
    if (k0 + BK < K) {
      __builtin_prefetch(&A[(size_t)(arow0 + ar) * lda + k0 + BK + ac], 0, 1);
      __builtin_prefetch(&Bm[(size_t)(k0 + BK + bkh) * ldb + bcol0 + bn], 0, 1);
    }
    __syncthreads();

    // A fragment: lane m = lane&15; K 0-7/16-23 (lo lanes) or 8-15/24-31 (hi lanes)
    const int arow = wm * 16 + (lane & 15);
    const int ah   = (lane >> 4) * 8;
    FragU fa;
    fa.q[0] = *(const uint4*)&sA[arow * BKP + ah];
    fa.q[1] = *(const uint4*)&sA[arow * BKP + ah + 16];

    // Load ALL four B fragments first, then issue 4 back-to-back WMMAs
    FragU fb[4];
#pragma unroll
    for (int j = 0; j < 4; ++j) {
      const int nn = wn * 64 + j * 16 + (lane & 15);
      const int bk = (lane >> 4) * 8;
      fb[j].q[0] = *(const uint4*)&sBT[nn * BKP + bk];
      fb[j].q[1] = *(const uint4*)&sBT[nn * BKP + bk + 16];
    }
#pragma unroll
    for (int j = 0; j < 4; ++j) {
      acc[j] = __builtin_amdgcn_wmma_f32_16x16x32_bf16(
          false, fa.v, false, fb[j].v, (short)0, acc[j], false, false);
    }
    __syncthreads();
  }
}

// C/D layout: VGPR r -> M = r (lanes 0-15), M = 8+r (lanes 16-31); N = lane&15
#define EPI_SETUP                                                   \
  const int lane_  = threadIdx.x & 31;                              \
  const int wid_   = threadIdx.x >> 5;                              \
  const int wm_    = wid_ >> 1;                                     \
  const int wn_    = wid_ & 1;                                      \
  const int rowin_ = wm_ * 16 + (lane_ >> 4) * 8;                   \
  const int cl_    = lane_ & 15;

// ---------------------------------------------------------------------------
// Grid-wide barrier for the persistent scan kernel (monotonic counter).
// ---------------------------------------------------------------------------
__device__ __forceinline__ void gridbar(unsigned* cnt, unsigned epoch) {
  __threadfence();
  __syncthreads();
  if (threadIdx.x == 0) {
    __hip_atomic_fetch_add(cnt, 1u, __ATOMIC_ACQ_REL, __HIP_MEMORY_SCOPE_AGENT);
    const unsigned target = epoch * (unsigned)NWG_SCAN;
    while (__hip_atomic_load(cnt, __ATOMIC_ACQUIRE, __HIP_MEMORY_SCOPE_AGENT) < target)
      __builtin_amdgcn_s_sleep(2);
  }
  __syncthreads();
  __threadfence();
}

// ---------------------------------------------------------------------------
// Prep kernels
// ---------------------------------------------------------------------------
__global__ void k_bt(const float* __restrict__ x, const float* __restrict__ fi,
                     unsigned short* __restrict__ bt) {
  size_t i = (size_t)blockIdx.x * blockDim.x + threadIdx.x;
  if (i >= (size_t)MROWS * DIN) return;
  size_t m = i / DIN, d = i - m * DIN;
  float v = x[i] * (1.f - fi[(m * 2 + 1) * DIN + d]);
  bt[i] = f2bf(v);
}

__global__ void k_dt(const float* __restrict__ fo, const int* __restrict__ mask,
                     unsigned short* __restrict__ dt) {
  size_t i = (size_t)blockIdx.x * blockDim.x + threadIdx.x;
  if (i >= (size_t)MROWS * DDISC) return;
  size_t m = i >> 9; int j = (int)(i & 511);
  dt[i] = f2bf(fo[(m * 2 + 1) * DIN + mask[j]]);
}

__global__ void k_cvt(const float* __restrict__ s, unsigned short* __restrict__ d, int n) {
  int i = blockIdx.x * blockDim.x + threadIdx.x;
  if (i < n) d[i] = f2bf(s[i]);
}

__global__ void k_zero_h(float* __restrict__ hf, unsigned short* __restrict__ hb,
                         unsigned* __restrict__ barcnt) {
  int i = blockIdx.x * blockDim.x + threadIdx.x;
  if (i < B_BATCH * HDIM) { hf[i] = 0.f; hb[i] = 0; }
  if (i == 0) *barcnt = 0u;
}

// ---------------------------------------------------------------------------
// Batched projections: pf/ps/pi = bt @ {W_fb,W_sb,W_ib} + bias   (time-major out)
// grid = (128, 24): y/8 selects matrix, y%8 selects 128-col slab
// ---------------------------------------------------------------------------
__global__ __launch_bounds__(256) void k_proj(
    const unsigned short* __restrict__ bt,
    const unsigned short* __restrict__ Wf, const unsigned short* __restrict__ Ws,
    const unsigned short* __restrict__ Wi,
    const float* __restrict__ ef, const float* __restrict__ es, const float* __restrict__ ei,
    float* __restrict__ pf, float* __restrict__ ps, float* __restrict__ pi)
{
  const int bm = blockIdx.x;
  const int gn = blockIdx.y;
  const int mat = gn >> 3;
  const int bcol0 = (gn & 7) * BN;
  const unsigned short* W = (mat == 0) ? Wf : (mat == 1) ? Ws : Wi;
  const float* e = (mat == 0) ? ef : (mat == 1) ? es : ei;
  float* outp = (mat == 0) ? pf : (mat == 1) ? ps : pi;

  v8f acc[4];
  gemm_core(bt, DIN, bm * BM, W, HDIM, bcol0, DIN, acc);

  EPI_SETUP
#pragma unroll
  for (int j = 0; j < 4; ++j)
#pragma unroll
    for (int r = 0; r < 8; ++r) {
      int m = bm * BM + rowin_ + r;
      int n = bcol0 + wn_ * 64 + j * 16 + cl_;
      int b = m >> 7, t = m & 127;            // m = b*T + t
      outp[(size_t)(t * B_BATCH + b) * HDIM + n] = acc[j][r] + e[n];
    }
}

// At = dt @ E   (time-major out)
__global__ __launch_bounds__(256) void k_At(
    const unsigned short* __restrict__ dt, const unsigned short* __restrict__ Ebf,
    float* __restrict__ At)
{
  const int bm = blockIdx.x;
  const int bcol0 = blockIdx.y * BN;
  v8f acc[4];
  gemm_core(dt, DDISC, bm * BM, Ebf, DIN, bcol0, DDISC, acc);
  EPI_SETUP
#pragma unroll
  for (int j = 0; j < 4; ++j)
#pragma unroll
    for (int r = 0; r < 8; ++r) {
      int m = bm * BM + rowin_ + r;
      int n = bcol0 + wn_ * 64 + j * 16 + cl_;
      int b = m >> 7, t = m & 127;
      At[(size_t)(t * B_BATCH + b) * DIN + n] = acc[j][r];
    }
}

// ---------------------------------------------------------------------------
// Persistent recurrent scan: one launch, 16 co-resident workgroups, internal
// t-loop, 3 dependent phases per step separated by grid barriers.
//   Phase 1 (16 WGs): f = sig(pf_t + h@W_fa), s = sig(ps_t + h@W_sa), hs = h*s
//   Phase 2 ( 8 WGs): c = tanh(pi_t + hs@W_ia), h = h*(1-f) + f*c
//   Phase 3 (16 WGs): q = h@W_ba, out[b][t] = sig(q*(1+At_t) + e_o)
// Weights (10 MB bf16 total) stay resident in L2 across all 128 steps.
// ---------------------------------------------------------------------------
__global__ __launch_bounds__(256) void k_scan(
    unsigned short* __restrict__ hb, float* __restrict__ hf,
    unsigned short* __restrict__ hs, float* __restrict__ f_ws,
    const unsigned short* __restrict__ Wfa, const unsigned short* __restrict__ Wsa,
    const unsigned short* __restrict__ Wia, const unsigned short* __restrict__ Wba,
    const float* __restrict__ pf, const float* __restrict__ ps,
    const float* __restrict__ pi, const float* __restrict__ At,
    const float* __restrict__ eo, float* __restrict__ out,
    unsigned* __restrict__ barcnt)
{
  const int g = blockIdx.x;          // 0..15
  unsigned epoch = 0;

  for (int t = 0; t < T_SEQ; ++t) {
    // ---- Phase 1: gates f (WGs 0-7) and s (WGs 8-15) ----------------------
    {
      const int isS = g >> 3;
      const int bcol0 = (g & 7) * BN;
      const unsigned short* W = isS ? Wsa : Wfa;
      v8f acc[4];
      gemm_core(hb, HDIM, 0, W, HDIM, bcol0, HDIM, acc);
      EPI_SETUP
#pragma unroll
      for (int j = 0; j < 4; ++j)
#pragma unroll
        for (int r = 0; r < 8; ++r) {
          int b = rowin_ + r;
          int n = bcol0 + wn_ * 64 + j * 16 + cl_;
          size_t hi = (size_t)b * HDIM + n;
          size_t pidx = (size_t)(t * B_BATCH + b) * HDIM + n;
          if (!isS) {
            f_ws[hi] = sigmoidf_(pf[pidx] + acc[j][r]);
          } else {
            float s = sigmoidf_(ps[pidx] + acc[j][r]);
            hs[hi] = f2bf(hf[hi] * s);
          }
        }
    }
    gridbar(barcnt, ++epoch);

    // ---- Phase 2: candidate + h update (WGs 0-7) --------------------------
    if (g < 8) {
      const int bcol0 = g * BN;
      v8f acc[4];
      gemm_core(hs, HDIM, 0, Wia, HDIM, bcol0, HDIM, acc);
      EPI_SETUP
#pragma unroll
      for (int j = 0; j < 4; ++j)
#pragma unroll
        for (int r = 0; r < 8; ++r) {
          int b = rowin_ + r;
          int n = bcol0 + wn_ * 64 + j * 16 + cl_;
          size_t hi = (size_t)b * HDIM + n;
          float c = tanhf(pi[(size_t)(t * B_BATCH + b) * HDIM + n] + acc[j][r]);
          float f = f_ws[hi];
          float hn = hf[hi] * (1.f - f) + f * c;
          hf[hi] = hn;
          hb[hi] = f2bf(hn);
        }
    }
    gridbar(barcnt, ++epoch);

    // ---- Phase 3: output projection (16 WGs) ------------------------------
    {
      const int bcol0 = g * BN;
      v8f acc[4];
      gemm_core(hb, HDIM, 0, Wba, DIN, bcol0, HDIM, acc);
      EPI_SETUP
#pragma unroll
      for (int j = 0; j < 4; ++j)
#pragma unroll
        for (int r = 0; r < 8; ++r) {
          int b = rowin_ + r;
          int n = bcol0 + wn_ * 64 + j * 16 + cl_;
          float a = At[(size_t)(t * B_BATCH + b) * DIN + n];
          float p = sigmoidf_(acc[j][r] * (1.f + a) + eo[n]);
          out[((size_t)b * T_SEQ + t) * DIN + n] = p;
        }
    }
    gridbar(barcnt, ++epoch);   // protects hs/f_ws from next step's phase 1
  }
}

// ---------------------------------------------------------------------------
extern "C" void kernel_launch(void* const* d_in, const int* in_sizes, int n_in,
                              void* d_out, int out_size, void* d_ws, size_t ws_size,
                              hipStream_t stream)
{
  (void)in_sizes; (void)n_in; (void)out_size; (void)ws_size;
  const float* x    = (const float*)d_in[0];
  const float* fo   = (const float*)d_in[1];
  const float* fi   = (const float*)d_in[2];
  const float* W_fb = (const float*)d_in[3];
  const float* W_fa = (const float*)d_in[4];
  const float* e_f  = (const float*)d_in[5];
  const float* W_ib = (const float*)d_in[6];
  const float* W_ia = (const float*)d_in[7];
  const float* e_i  = (const float*)d_in[8];
  const float* W_sb = (const float*)d_in[9];
  const float* W_sa = (const float*)d_in[10];
  const float* e_s  = (const float*)d_in[11];
  const float* W_ba = (const float*)d_in[12];
  const float* e_o  = (const float*)d_in[13];
  const float* E    = (const float*)d_in[14];
  const int*   mask = (const int*)d_in[15];
  float* out = (float*)d_out;

  char* wptr = (char*)d_ws;
  auto alloc = [&](size_t bytes) -> void* {
    void* p = (void*)wptr;
    wptr += (bytes + 255) & ~(size_t)255;
    return p;
  };
  unsigned short* bt   = (unsigned short*)alloc((size_t)MROWS * DIN * 2);
  unsigned short* dt   = (unsigned short*)alloc((size_t)MROWS * DDISC * 2);
  unsigned short* bWfb = (unsigned short*)alloc((size_t)DIN * HDIM * 2);
  unsigned short* bWsb = (unsigned short*)alloc((size_t)DIN * HDIM * 2);
  unsigned short* bWib = (unsigned short*)alloc((size_t)DIN * HDIM * 2);
  unsigned short* bWfa = (unsigned short*)alloc((size_t)HDIM * HDIM * 2);
  unsigned short* bWsa = (unsigned short*)alloc((size_t)HDIM * HDIM * 2);
  unsigned short* bWia = (unsigned short*)alloc((size_t)HDIM * HDIM * 2);
  unsigned short* bWba = (unsigned short*)alloc((size_t)HDIM * DIN * 2);
  unsigned short* bE   = (unsigned short*)alloc((size_t)DDISC * DIN * 2);
  float* pf = (float*)alloc((size_t)T_SEQ * B_BATCH * HDIM * 4);
  float* ps = (float*)alloc((size_t)T_SEQ * B_BATCH * HDIM * 4);
  float* pi = (float*)alloc((size_t)T_SEQ * B_BATCH * HDIM * 4);
  float* At = (float*)alloc((size_t)T_SEQ * B_BATCH * DIN * 4);
  float*          hf  = (float*)alloc((size_t)B_BATCH * HDIM * 4);
  unsigned short* hb  = (unsigned short*)alloc((size_t)B_BATCH * HDIM * 2);
  unsigned short* hs  = (unsigned short*)alloc((size_t)B_BATCH * HDIM * 2);
  float*          fws = (float*)alloc((size_t)B_BATCH * HDIM * 4);
  unsigned*       bar = (unsigned*)alloc(256);

  // --- prep: pack activations + weights to bf16, zero h + barrier ---------
  {
    size_t n = (size_t)MROWS * DIN;
    k_bt<<<dim3((unsigned)((n + 255) / 256)), 256, 0, stream>>>(x, fi, bt);
  }
  {
    size_t n = (size_t)MROWS * DDISC;
    k_dt<<<dim3((unsigned)((n + 255) / 256)), 256, 0, stream>>>(fo, mask, dt);
  }
  auto cvt = [&](const float* s, unsigned short* d, int n) {
    k_cvt<<<dim3((n + 255) / 256), 256, 0, stream>>>(s, d, n);
  };
  cvt(W_fb, bWfb, DIN * HDIM);
  cvt(W_sb, bWsb, DIN * HDIM);
  cvt(W_ib, bWib, DIN * HDIM);
  cvt(W_fa, bWfa, HDIM * HDIM);
  cvt(W_sa, bWsa, HDIM * HDIM);
  cvt(W_ia, bWia, HDIM * HDIM);
  cvt(W_ba, bWba, HDIM * DIN);
  cvt(E,    bE,   DDISC * DIN);
  k_zero_h<<<dim3((B_BATCH * HDIM) / 256), 256, 0, stream>>>(hf, hb, bar);

  // --- batched WMMA GEMMs --------------------------------------------------
  k_proj<<<dim3(MROWS / BM, 24), 256, 0, stream>>>(bt, bWfb, bWsb, bWib,
                                                   e_f, e_s, e_i, pf, ps, pi);
  k_At<<<dim3(MROWS / BM, DIN / BN), 256, 0, stream>>>(dt, bE, At);

  // --- recurrent scan: single persistent cooperative kernel ----------------
  k_scan<<<dim3(NWG_SCAN), 256, 0, stream>>>(hb, hf, hs, fws,
                                             bWfa, bWsa, bWia, bWba,
                                             pf, ps, pi, At, e_o, out, bar);
}
